// GAT_64639257805503
// MI455X (gfx1250) — matrix operand order repaired and checked by reading, not compiled, not run on hardware
//
#include <hip/hip_runtime.h>

// ---------------------------------------------------------------------------
// GAT (3-layer) for MI455X / gfx1250.
// GEMMs via v_wmma_f32_16x16x32_bf16 (f32 accumulate), W pre-swizzled in LDS
// so each B fragment is a single 32B ds_load; edge segment-softmax + scatter
// via f32 global atomics (working set is L2-resident: 192MB global L2).
// ---------------------------------------------------------------------------

typedef __attribute__((ext_vector_type(16))) __bf16 v16bf;
typedef __attribute__((ext_vector_type(8)))  float  v8f;

__device__ __forceinline__ unsigned short f32_to_bf16(float f) {
  unsigned int u = __float_as_uint(f);
  unsigned int r = u + 0x7FFFu + ((u >> 16) & 1u);   // round-to-nearest-even
  return (unsigned short)(r >> 16);
}

// order-preserving float <-> uint map for atomicMax on floats
__device__ __forceinline__ unsigned int ford_enc(float f) {
  unsigned int i = __float_as_uint(f);
  return (i & 0x80000000u) ? ~i : (i | 0x80000000u);
}
__device__ __forceinline__ float ford_dec(unsigned int u) {
  return __uint_as_float((u & 0x80000000u) ? (u ^ 0x80000000u) : ~u);
}
// encoding of -inf (init value for the segment max)
#define FORD_NEG_INF 0x007FFFFFu

// ---------------------------------------------------------------------------
// GEMM: H[n, COUT] = A[n, CIN] @ W[CIN, COUT], bf16 WMMA, f32 accum.
// One wave computes a 16 x COUT tile; 8 waves per 256-thread block.
// W is staged once per block into LDS *pre-swizzled into WMMA B-fragment
// layout*, so the inner loop reads each 16-half fragment as one b128-pair.
//
// CDNA5 16-bit B layout (32x16, K x N), per 32-K step ks and 16-N tile t:
//   lane 0-15 : K = ks*32 + (j&7) + 16*(j>>3)      (j = element 0..15), N = t*16 + lane
//   lane 16-31: K = ks*32 + 8 + (j&7) + 16*(j>>3),  N = t*16 + (lane-16)
// Swizzled LDS index: ((ks*NT + t)*32 + lane)*16 + j.
// ---------------------------------------------------------------------------
template <int CIN, int COUT>
__global__ __launch_bounds__(256) void gemm_wmma(
    const float* __restrict__ A, const float* __restrict__ W,
    float* __restrict__ H, int n) {
  constexpr int NT = COUT / 16;      // n-tiles
  constexpr int KS = CIN / 32;       // k-steps
  constexpr int WE = CIN * COUT;     // total W elements
  __shared__ alignas(32) unsigned short sWs[WE];

  const int tid = threadIdx.x;
  // stage W -> LDS, swizzled into fragment layout
  for (int idx = tid; idx < WE; idx += 256) {
    const int j    = idx & 15;
    const int ln   = (idx >> 4) & 31;
    const int frag = idx >> 9;              // ks*NT + t
    const int t    = frag % NT;
    const int ks   = frag / NT;
    const int k    = ks * 32 + ((ln >> 4) << 3) + (j & 7) + ((j >> 3) << 4);
    const int ncol = t * 16 + (ln & 15);
    sWs[idx] = f32_to_bf16(W[k * COUT + ncol]);
  }
  __syncthreads();

  const int lane = tid & 31;
  const int wave = tid >> 5;
  const int rowTiles = (n + 15) >> 4;
  const int rowTile = blockIdx.x * 8 + wave;
  if (rowTile >= rowTiles) return;
  const int rowBase = rowTile << 4;
  const int laneHi = lane >> 4;   // 0: lanes 0-15, 1: lanes 16-31
  const int lane15 = lane & 15;

  v8f acc[NT];
  v8f zero = {};
#pragma unroll
  for (int t = 0; t < NT; ++t) acc[t] = zero;

  const float* __restrict__ arow = A + (size_t)(rowBase + lane15) * CIN;
  const unsigned short* __restrict__ fragBase = sWs + lane * 16;

#pragma unroll
  for (int ks = 0; ks < KS; ++ks) {
    // A fragment: lanes 0-15 hold K in {kb..kb+7, kb+16..kb+23},
    //             lanes 16-31 hold K in {kb+8..kb+15, kb+24..kb+31}.
    const int kb = ks * 32;
    const int k0 = kb + laneHi * 8;
    float4 a0 = *(const float4*)(arow + k0);
    float4 a1 = *(const float4*)(arow + k0 + 4);
    float4 a2 = *(const float4*)(arow + k0 + 16);
    float4 a3 = *(const float4*)(arow + k0 + 20);
    union { v16bf v; unsigned short s[16]; } av;
    av.s[0]  = f32_to_bf16(a0.x); av.s[1]  = f32_to_bf16(a0.y);
    av.s[2]  = f32_to_bf16(a0.z); av.s[3]  = f32_to_bf16(a0.w);
    av.s[4]  = f32_to_bf16(a1.x); av.s[5]  = f32_to_bf16(a1.y);
    av.s[6]  = f32_to_bf16(a1.z); av.s[7]  = f32_to_bf16(a1.w);
    av.s[8]  = f32_to_bf16(a2.x); av.s[9]  = f32_to_bf16(a2.y);
    av.s[10] = f32_to_bf16(a2.z); av.s[11] = f32_to_bf16(a2.w);
    av.s[12] = f32_to_bf16(a3.x); av.s[13] = f32_to_bf16(a3.y);
    av.s[14] = f32_to_bf16(a3.z); av.s[15] = f32_to_bf16(a3.w);

#pragma unroll
    for (int t = 0; t < NT; ++t) {
      const v16bf bfrag =
          *(const v16bf*)(fragBase + (size_t)(ks * NT + t) * 512);
      acc[t] = __builtin_amdgcn_wmma_f32_16x16x32_bf16(
          false, av.v, false, bfrag, (short)0, acc[t], false, false);
    }
  }

  // C/D layout: VGPR r -> M = r (+8 for lanes 16-31), N = lane&15.
#pragma unroll
  for (int t = 0; t < NT; ++t) {
    const int ncol = t * 16 + lane15;
#pragma unroll
    for (int r = 0; r < 8; ++r) {
      const int m = rowBase + r + laneHi * 8;
      H[(size_t)m * COUT + ncol] = acc[t][r];
    }
  }
}

// ---------------------------------------------------------------------------
// Per-node attention scores: a_s[i] = dot(h[i,:], asrc), a_d[i] = dot(h[i,:], adst)
// C/4 lanes per node (8 or 16), float4 loads, shfl_xor reduction.
// ---------------------------------------------------------------------------
__global__ __launch_bounds__(256) void attn_scores(
    const float* __restrict__ h, const float* __restrict__ av,
    const float* __restrict__ dv, float* __restrict__ a_s,
    float* __restrict__ a_d, long long n, int C) {
  const int G = C >> 2;
  long long gid = (long long)blockIdx.x * 256 + threadIdx.x;
  long long node = gid / G;
  int sub = (int)(gid - node * (long long)G);
  if (node >= n) return;
  float4 hv = *(const float4*)(h + node * C + sub * 4);
  float4 s4 = *(const float4*)(av + sub * 4);
  float4 d4 = *(const float4*)(dv + sub * 4);
  float ss = hv.x * s4.x + hv.y * s4.y + hv.z * s4.z + hv.w * s4.w;
  float dd = hv.x * d4.x + hv.y * d4.y + hv.z * d4.z + hv.w * d4.w;
  for (int o = G >> 1; o > 0; o >>= 1) {
    ss += __shfl_xor(ss, o, 32);
    dd += __shfl_xor(dd, o, 32);
  }
  if (sub == 0) { a_s[node] = ss; a_d[node] = dd; }
}

// ---------------------------------------------------------------------------
__global__ __launch_bounds__(256) void init_layer(
    float* __restrict__ agg, unsigned int* __restrict__ emx,
    float* __restrict__ den, long long n, int C) {
  long long gid = (long long)blockIdx.x * 256 + threadIdx.x;
  if (gid < n * (long long)C) agg[gid] = 0.0f;
  if (gid < n) { emx[gid] = FORD_NEG_INF; den[gid] = 0.0f; }
}

// pass 1: e = leaky(a_s[src]+a_d[dst], 0.2); segment max over dst
__global__ __launch_bounds__(256) void edge_max(
    const int* __restrict__ src, const int* __restrict__ dst,
    const float* __restrict__ a_s, const float* __restrict__ a_d,
    unsigned int* __restrict__ emx, long long E, long long n) {
  long long e = (long long)blockIdx.x * 256 + threadIdx.x;
  if (e >= E + n) return;
  int s, d;
  if (e < E) { s = src[e]; d = dst[e]; } else { s = d = (int)(e - E); }
  float el = a_s[s] + a_d[d];
  el = (el > 0.0f) ? el : 0.2f * el;
  atomicMax(&emx[d], ford_enc(el));
}

// pass 2: ex = exp(e - emax[dst]); store numerator, segment-sum denom
__global__ __launch_bounds__(256) void edge_expsum(
    const int* __restrict__ src, const int* __restrict__ dst,
    const float* __restrict__ a_s, const float* __restrict__ a_d,
    const unsigned int* __restrict__ emx, float* __restrict__ den,
    float* __restrict__ exf, long long E, long long n) {
  long long e = (long long)blockIdx.x * 256 + threadIdx.x;
  if (e >= E + n) return;
  int s, d;
  if (e < E) { s = src[e]; d = dst[e]; } else { s = d = (int)(e - E); }
  float el = a_s[s] + a_d[d];
  el = (el > 0.0f) ? el : 0.2f * el;
  float mx = ford_dec(emx[d]);
  if (!(mx > -1e38f && mx < 1e38f)) mx = 0.0f;   // reference: non-finite -> 0
  float ex = __expf(el - mx);
  exf[e] = ex;
  atomicAdd(&den[d], ex);
}

// pass 3: agg[dst,:] += (ex/denom[dst]) * h[src,:]; C/4 lanes per edge
__global__ __launch_bounds__(256) void edge_aggregate(
    const int* __restrict__ src, const int* __restrict__ dst,
    const float* __restrict__ exf, const float* __restrict__ den,
    const float* __restrict__ h, float* __restrict__ agg,
    long long E, long long n, int C) {
  const int G = C >> 2;
  long long gid = (long long)blockIdx.x * 256 + threadIdx.x;
  long long e = gid / G;
  int sub = (int)(gid - e * (long long)G);
  if (e >= E + n) return;
  int s, d;
  if (e < E) { s = src[e]; d = dst[e]; } else { s = d = (int)(e - E); }
  float alpha = exf[e] / (den[d] + 1e-16f);
  float4 hv = *(const float4*)(h + (size_t)s * C + sub * 4);
  float* o = agg + (size_t)d * C + sub * 4;
  atomicAdd(o + 0, alpha * hv.x);
  atomicAdd(o + 1, alpha * hv.y);
  atomicAdd(o + 2, alpha * hv.z);
  atomicAdd(o + 3, alpha * hv.w);
}

// bias + optional leakyReLU(0.01)
__global__ __launch_bounds__(256) void finalize(
    const float* __restrict__ agg, const float* __restrict__ b,
    float* __restrict__ out, long long n, int C, int act) {
  long long gid = (long long)blockIdx.x * 256 + threadIdx.x;
  if (gid >= n * (long long)C) return;
  int c = (int)(gid % C);
  float v = agg[gid] + b[c];
  if (act) v = (v > 0.0f) ? v : 0.01f * v;
  out[gid] = v;
}

// ---------------------------------------------------------------------------
static inline int cdivll(long long a, long long b) { return (int)((a + b - 1) / b); }

extern "C" void kernel_launch(void* const* d_in, const int* in_sizes, int n_in,
                              void* d_out, int out_size, void* d_ws, size_t ws_size,
                              hipStream_t stream) {
  (void)n_in; (void)out_size; (void)ws_size;
  const float* x   = (const float*)d_in[0];
  const int*   ei  = (const int*)d_in[1];
  const float* W1  = (const float*)d_in[2];
  const float* as1 = (const float*)d_in[3];
  const float* ad1 = (const float*)d_in[4];
  const float* b1  = (const float*)d_in[5];
  const float* W2  = (const float*)d_in[6];
  const float* as2 = (const float*)d_in[7];
  const float* ad2 = (const float*)d_in[8];
  const float* b2  = (const float*)d_in[9];
  const float* W3  = (const float*)d_in[10];
  const float* as3 = (const float*)d_in[11];
  const float* ad3 = (const float*)d_in[12];
  const float* b3  = (const float*)d_in[13];

  const long long n = in_sizes[0] / 256;   // N nodes (x is [N, 256])
  const long long E = in_sizes[1] / 2;     // edge_index is [2, E]
  const int* src = ei;
  const int* dst = ei + E;

  // workspace carve-out
  char* ws = (char*)d_ws;
  size_t off = 0;
  auto carve = [&](size_t bytes) {
    void* p = ws + off;
    off += (bytes + 255) & ~(size_t)255;
    return p;
  };
  float*        h   = (float*)carve((size_t)n * 64 * 4);
  float*        act = (float*)carve((size_t)n * 64 * 4);
  float*        agg = (float*)carve((size_t)n * 64 * 4);
  float*        a_s = (float*)carve((size_t)n * 4);
  float*        a_d = (float*)carve((size_t)n * 4);
  unsigned int* emx = (unsigned int*)carve((size_t)n * 4);
  float*        den = (float*)carve((size_t)n * 4);
  float*        exf = (float*)carve((size_t)(E + n) * 4);

  const long long T = E + n;   // edges incl. self loops
  const int rowTiles = (int)((n + 15) >> 4);
  const int gemmBlocks = (rowTiles + 7) / 8;

  // attention + segment-softmax + aggregate + bias/activation for one layer
  auto edge_phase = [&](const float* av, const float* dv, const float* b,
                        int cout, float* out, int actflag) {
    const int G = cout / 4;
    attn_scores<<<cdivll(n * G, 256), 256, 0, stream>>>(h, av, dv, a_s, a_d, n, cout);
    init_layer<<<cdivll(n * cout, 256), 256, 0, stream>>>(agg, emx, den, n, cout);
    edge_max<<<cdivll(T, 256), 256, 0, stream>>>(src, dst, a_s, a_d, emx, E, n);
    edge_expsum<<<cdivll(T, 256), 256, 0, stream>>>(src, dst, a_s, a_d, emx, den, exf, E, n);
    edge_aggregate<<<cdivll(T * G, 256), 256, 0, stream>>>(src, dst, exf, den, h, agg, E, n, cout);
    finalize<<<cdivll(n * cout, 256), 256, 0, stream>>>(agg, b, out, n, cout, actflag);
  };

  // layer 1: x[ n,256 ] @ W1 -> 64, leaky 0.01
  gemm_wmma<256, 64><<<gemmBlocks, 256, 0, stream>>>(x, W1, h, (int)n);
  edge_phase(as1, ad1, b1, 64, act, 1);
  // layer 2: act[ n,64 ] @ W2 -> 64, leaky 0.01
  gemm_wmma<64, 64><<<gemmBlocks, 256, 0, stream>>>(act, W2, h, (int)n);
  edge_phase(as2, ad2, b2, 64, act, 1);
  // layer 3: act[ n,64 ] @ W3 -> 32, no activation (mean over 1 head = identity)
  gemm_wmma<64, 32><<<gemmBlocks, 256, 0, stream>>>(act, W3, h, (int)n);
  edge_phase(as3, ad3, b3, 32, (float*)d_out, 0);
}
